// MultiHeadedAttention_64501818851578
// MI455X (gfx1250) — matrix operand order repaired and checked by reading, not compiled, not run on hardware
//
#include <hip/hip_runtime.h>
#include <hip/hip_bf16.h>

// ---------------------------------------------------------------------------
// Problem constants (from reference)
// ---------------------------------------------------------------------------
#define B_DIM   128
#define S_DIM   256
#define DM      128      // D_MODEL
#define H_DIM   8
#define DK      16       // D_MODEL / H
#define KSZ     7
#define NTOP    30
#define SAMPK   30
#define BS_ELTS (S_DIM * DM)          // 32768 elements per batch (S x D_MODEL)
#define PAD_ROWS 134                  // 6 zero pad rows + 128 feature rows

typedef __attribute__((ext_vector_type(16))) _Float16 v16h;
typedef __attribute__((ext_vector_type(8)))  float    v8f;
typedef int v4i __attribute__((vector_size(16)));     // matches builtin param

union FragH {                 // one WMMA f16 operand (8 VGPRs = 16 halves/lane)
    v16h h;
    uint4 q[2];
};

// ---------------------------------------------------------------------------
// Async global->LDS copy (gfx1250 GLOBAL_LOAD_ASYNC_TO_LDS_B128), guarded so
// the file always compiles; falls back to a plain 16B copy if unavailable.
// Builtin signature (from clang diagnostic): arg0 = v4i addrspace(1)*,
// arg1 = v4i addrspace(3)*, arg2 = i32 offset, arg3 = i32 cpol.
// ---------------------------------------------------------------------------
#if __has_builtin(__builtin_amdgcn_global_load_async_to_lds_b128) && \
    __has_builtin(__builtin_amdgcn_s_wait_asynccnt)
#define USE_ASYNC_LDS 1
#else
#define USE_ASYNC_LDS 0
#endif

__device__ __forceinline__ void async_copy16(void* lds_dst, const void* gsrc)
{
#if USE_ASYNC_LDS
    __builtin_amdgcn_global_load_async_to_lds_b128(
        (__attribute__((address_space(1))) v4i*)(void*)gsrc,
        (__attribute__((address_space(3))) v4i*)lds_dst,
        0, 0);
#else
    *(uint4*)lds_dst = *(const uint4*)gsrc;
#endif
}

__device__ __forceinline__ void async_wait_all()
{
#if USE_ASYNC_LDS
    __builtin_amdgcn_s_wait_asynccnt(0);
#endif
}

// ---------------------------------------------------------------------------
// Kernel 1: rotary embedding for q and k (f32 -> f32 workspace)
// ---------------------------------------------------------------------------
__global__ __launch_bounds__(256) void rotary_kernel(
    const float* __restrict__ q, const float* __restrict__ k,
    float* __restrict__ qr, float* __restrict__ kr)
{
    int gid = blockIdx.x * 256 + threadIdx.x;          // 0 .. B*S*64-1
    int p   = gid & 63;
    int s   = (gid >> 6) & (S_DIM - 1);
    int b   = gid >> 14;
    size_t base = (size_t)b * BS_ELTS + (size_t)s * DM + 2 * p;

    float freq = __powf(10000.0f, -((float)(2 * p)) * (1.0f / (float)DM));
    float ang  = (float)s * freq;
    float c = __cosf(ang), sn = __sinf(ang);

    float q0 = q[base], q1 = q[base + 1];
    qr[base]     = q0 * c - q1 * sn;
    qr[base + 1] = q0 * sn + q1 * c;

    float k0 = k[base], k1 = k[base + 1];
    kr[base]     = k0 * c - k1 * sn;
    kr[base + 1] = k0 * sn + k1 * c;
}

// ---------------------------------------------------------------------------
// Kernel 2a: repack conv weights (O,I,K) f32  ->  [t][o][i] f16
// ---------------------------------------------------------------------------
__global__ __launch_bounds__(256) void convert_conv_w_kernel(
    const float* __restrict__ w, _Float16* __restrict__ wh)
{
    int gid = blockIdx.x * 256 + threadIdx.x;          // t*65536 + o*256 + i
    if (gid >= KSZ * S_DIM * S_DIM) return;
    int t   = gid >> 16;
    int rem = gid & 65535;
    int o   = rem >> 8;
    int i   = rem & 255;
    wh[gid] = (_Float16)w[((size_t)o * S_DIM + i) * KSZ + t];
}

// Kernel 2b: lin_w f32 -> f16 (same [dm][d] layout)
__global__ __launch_bounds__(256) void convert_lin_kernel(
    const float* __restrict__ w, _Float16* __restrict__ wh)
{
    int gid = blockIdx.x * 256 + threadIdx.x;
    if (gid >= DM * DM) return;
    wh[gid] = (_Float16)w[gid];
}

// ---------------------------------------------------------------------------
// Kernel 2c: build padded transposed f16 conv input  xt[b][r][i]
//   r = 0..5   -> zero (causal left pad)
//   r = 6..133 -> x[b][i][r-6]
// Makes each conv block's LDS tile one contiguous region for async copy.
// ---------------------------------------------------------------------------
__global__ __launch_bounds__(256) void transpose_pad_kernel(
    const float* __restrict__ x, _Float16* __restrict__ xt)
{
    int gid = blockIdx.x * 256 + threadIdx.x;
    if (gid >= B_DIM * PAD_ROWS * 256) return;
    int i = gid & 255;
    int r = (gid >> 8) % PAD_ROWS;
    int b = gid / (PAD_ROWS * 256);
    float v = 0.0f;
    if (r >= 6) v = x[(size_t)b * BS_ELTS + (size_t)i * DM + (r - 6)];
    xt[gid] = (_Float16)v;
}

// ---------------------------------------------------------------------------
// Kernel 3: conv as 7 shifted WMMA GEMMs.
//   y[b,o,d] = bias[o] + sum_{t,i} W[t][o][i] * x[b,i,d+t-6]
// Block: (dg, b) -> all 256 o rows, d cols dg*64..+63 of one batch.
// Each wave: 2 M-tiles (32 o rows) x 4 N-tiles (64 d cols).
// LDS tile = contiguous slice of padded transposed input, async-copied.
// ---------------------------------------------------------------------------
__global__ __launch_bounds__(256) void conv_gemm_kernel(
    const _Float16* __restrict__ xt, const _Float16* __restrict__ wh,
    const float* __restrict__ bias, float* __restrict__ y)
{
    __shared__ _Float16 Xt[70 * 256];                  // 35840 B, contiguous

    const int tid = threadIdx.x;
    const int dg  = blockIdx.x;                        // 0..1  (d half)
    const int b   = blockIdx.y;

    // contiguous async copy: padded rows dg*64 .. dg*64+69 of batch b
    const _Float16* src = xt + ((size_t)b * PAD_ROWS + dg * 64) * 256;
    for (int c = tid; c < 2240; c += 256)              // 2240 x 16B = 35840 B
        async_copy16(&Xt[c * 8], src + c * 8);
    async_wait_all();
    __syncthreads();

    const int wave = tid >> 5;
    const int lane = tid & 31;
    const int lrow = lane & 15;
    const int hi   = lane >> 4;                        // 0 or 1
    const int kbA  = hi ? 8 : 0;                       // A frag K base
    const int koB  = hi ? 8 : 0;                       // B frag K offset
    const int m0   = wave * 32;                        // first of 2 M tiles

    v8f acc0[4] = {};
    v8f acc1[4] = {};

    for (int t = 0; t < KSZ; ++t) {
        const _Float16* w0 = wh + ((size_t)t * S_DIM + m0 + lrow) * S_DIM;
        const _Float16* w1 = w0 + 16 * S_DIM;
        for (int kk = 0; kk < 256; kk += 32) {
            __builtin_prefetch(w0 + kk + 64, 0, 3);    // global_prefetch_b8
            FragH a0, a1;
            a0.q[0] = *reinterpret_cast<const uint4*>(w0 + kk + kbA);
            a0.q[1] = *reinterpret_cast<const uint4*>(w0 + kk + 16 + kbA);
            a1.q[0] = *reinterpret_cast<const uint4*>(w1 + kk + kbA);
            a1.q[1] = *reinterpret_cast<const uint4*>(w1 + kk + 16 + kbA);

            FragH bf[4];
#pragma unroll
            for (int nt = 0; nt < 4; ++nt) {
                const _Float16* bp = &Xt[(nt * 16 + lrow + t) * 256 + kk + koB];
                bf[nt].q[0] = *reinterpret_cast<const uint4*>(bp);
                bf[nt].q[1] = *reinterpret_cast<const uint4*>(bp + 16);
            }
#pragma unroll
            for (int nt = 0; nt < 4; ++nt) {
                acc0[nt] = __builtin_amdgcn_wmma_f32_16x16x32_f16(
                    false, a0.h, false, bf[nt].h, (short)0, acc0[nt], false, false);
                acc1[nt] = __builtin_amdgcn_wmma_f32_16x16x32_f16(
                    false, a1.h, false, bf[nt].h, (short)0, acc1[nt], false, false);
            }
        }
    }

    // bias + store (C layout: lane col N = lrow, row = r + 8*hi)
    float* yb = y + (size_t)b * BS_ELTS;
    float bv0[8], bv1[8];
#pragma unroll
    for (int r = 0; r < 8; ++r) {
        bv0[r] = bias[m0 + r + hi * 8];
        bv1[r] = bias[m0 + 16 + r + hi * 8];
    }
#pragma unroll
    for (int nt = 0; nt < 4; ++nt) {
        int col = dg * 64 + nt * 16 + lrow;
#pragma unroll
        for (int r = 0; r < 8; ++r) {
            int o0 = m0 + r + hi * 8;
            yb[(size_t)o0 * DM + col]        = acc0[nt][r] + bv0[r];
            yb[(size_t)(o0 + 16) * DM + col] = acc1[nt][r] + bv1[r];
        }
    }
}

// ---------------------------------------------------------------------------
// Kernel 4: sampled QK scoring + M = max - mean + iterative top-30.
// ---------------------------------------------------------------------------
__global__ __launch_bounds__(256) void topk_kernel(
    const float* __restrict__ qr, const float* __restrict__ kc,
    const int* __restrict__ isamp, int* __restrict__ mtop)
{
    __shared__ float Mv[256];
    __shared__ float rv[256];
    __shared__ int   ri[256];

    const int tid = threadIdx.x;
    const int bh  = blockIdx.x;
    const int b   = bh >> 3;
    const int h   = bh & 7;

    float qv[DK];
    const float* qrow = qr + (size_t)b * BS_ELTS + (size_t)tid * DM + h * DK;
#pragma unroll
    for (int j = 0; j < DK; ++j) qv[j] = qrow[j];

    const float* kb = kc + (size_t)b * BS_ELTS + h * DK;
    float mx = -3.0e38f, sm = 0.0f;
    for (int s = 0; s < SAMPK; ++s) {
        int idx = isamp[tid * SAMPK + s];
        const float* krow = kb + (size_t)idx * DM;
        float d = 0.0f;
#pragma unroll
        for (int j = 0; j < DK; ++j) d += qv[j] * krow[j];
        mx = fmaxf(mx, d);
        sm += d;
    }
    Mv[tid] = mx - sm * (1.0f / (float)SAMPK);
    __syncthreads();

    for (int n = 0; n < NTOP; ++n) {
        rv[tid] = Mv[tid]; ri[tid] = tid;
        __syncthreads();
        for (int off = 128; off > 0; off >>= 1) {
            if (tid < off && rv[tid + off] > rv[tid]) {
                rv[tid] = rv[tid + off];
                ri[tid] = ri[tid + off];
            }
            __syncthreads();
        }
        if (tid == 0) {
            int best = ri[0];
            mtop[bh * 32 + n] = best;
            Mv[best] = -3.0e38f;
        }
        __syncthreads();
    }
}

// ---------------------------------------------------------------------------
// Kernel 5: attention on top rows + context assembly -> f16 activation x.
// ---------------------------------------------------------------------------
__global__ __launch_bounds__(256) void attn_kernel(
    const float* __restrict__ qr, const float* __restrict__ kc,
    const float* __restrict__ vc, const int* __restrict__ mtop,
    _Float16* __restrict__ xh)
{
    __shared__ _Float16 ksh[256 * DK];
    __shared__ _Float16 vsh[256 * DK];
    __shared__ float attnsh[NTOP * 256];
    __shared__ float qred[NTOP * DK];
    __shared__ float outsh[NTOP * DK];
    __shared__ float vmean[DK];
    __shared__ float red[256];
    __shared__ int   sel[256];

    const int tid = threadIdx.x;
    const int bh  = blockIdx.x;
    const int b   = bh >> 3;
    const int h   = bh & 7;

    const float* kb = kc + (size_t)b * BS_ELTS + h * DK;
    const float* vb = vc + (size_t)b * BS_ELTS + h * DK;
#pragma unroll
    for (int j = 0; j < DK; ++j) {
        ksh[tid * DK + j] = (_Float16)kb[(size_t)tid * DM + j];
        vsh[tid * DK + j] = (_Float16)vb[(size_t)tid * DM + j];
    }
    sel[tid] = -1;
    __syncthreads();

    if (tid < NTOP) {
        int idx = mtop[bh * 32 + tid];
        sel[idx] = tid;
        const float* qrow = qr + (size_t)b * BS_ELTS + (size_t)idx * DM + h * DK;
        for (int j = 0; j < DK; ++j) qred[tid * DK + j] = qrow[j];
    }
    if (tid < DK) {
        float a = 0.0f;
        for (int l = 0; l < 256; ++l) a += (float)vsh[l * DK + tid];
        vmean[tid] = a * (1.0f / 256.0f);
    }
    __syncthreads();

    for (int u = 0; u < NTOP; ++u) {
        float sc = 0.0f;
#pragma unroll
        for (int j = 0; j < DK; ++j) sc += qred[u * DK + j] * (float)ksh[tid * DK + j];
        sc *= 0.25f;                                   // 1/sqrt(16)

        red[tid] = sc; __syncthreads();
        for (int off = 128; off > 0; off >>= 1) {
            if (tid < off) red[tid] = fmaxf(red[tid], red[tid + off]);
            __syncthreads();
        }
        float mxv = red[0]; __syncthreads();

        float e = __expf(sc - mxv);
        red[tid] = e; __syncthreads();
        for (int off = 128; off > 0; off >>= 1) {
            if (tid < off) red[tid] += red[tid + off];
            __syncthreads();
        }
        float ssum = red[0]; __syncthreads();

        attnsh[u * 256 + tid] = e / ssum;
    }
    __syncthreads();

    for (int p = tid; p < NTOP * DK; p += 256) {
        int u = p >> 4, j = p & 15;
        float a = 0.0f;
        for (int l = 0; l < 256; ++l)
            a += attnsh[u * 256 + l] * (float)vsh[l * DK + j];
        outsh[p] = a;
    }
    __syncthreads();

    int u = sel[tid];
    _Float16* xrow = xh + (size_t)(b * S_DIM + tid) * DM + h * DK;
#pragma unroll
    for (int j = 0; j < DK; ++j) {
        float val = (u >= 0) ? outsh[u * DK + j] : vmean[j];
        xrow[j] = (_Float16)val;
    }
}

// ---------------------------------------------------------------------------
// Kernel 6: final linear  out = x @ lin_w^T + b  via WMMA.
// lin_w (f16) is async-staged into LDS once per block.
// ---------------------------------------------------------------------------
__global__ __launch_bounds__(256) void linear_gemm_kernel(
    const _Float16* __restrict__ xh, const _Float16* __restrict__ wl,
    const float* __restrict__ lb, float* __restrict__ out)
{
    __shared__ _Float16 Wsh[DM * DM];                  // 32 KB

    const int tid = threadIdx.x;
    for (int c = tid; c < 2048; c += 256)              // 2048 x 16B = 32 KB
        async_copy16(&Wsh[c * 8], wl + c * 8);
    async_wait_all();
    __syncthreads();

    const int wave  = tid >> 5;
    const int lane  = tid & 31;
    const int lrow  = lane & 15;
    const int hi    = lane >> 4;
    const int kbA   = hi ? 8 : 0;
    const int koB   = hi ? 8 : 0;
    const int mBase = blockIdx.x * 128 + wave * 16;

    v8f acc[8] = {};
    const _Float16* arow = xh + (size_t)(mBase + lrow) * DM;

#pragma unroll
    for (int kk = 0; kk < DM; kk += 32) {
        FragH a;
        a.q[0] = *reinterpret_cast<const uint4*>(arow + kk + kbA);
        a.q[1] = *reinterpret_cast<const uint4*>(arow + kk + 16 + kbA);

        FragH bf[8];
#pragma unroll
        for (int nt = 0; nt < 8; ++nt) {
            const _Float16* bp = &Wsh[(nt * 16 + lrow) * DM + kk + koB];
            bf[nt].q[0] = *reinterpret_cast<const uint4*>(bp);
            bf[nt].q[1] = *reinterpret_cast<const uint4*>(bp + 16);
        }
#pragma unroll
        for (int nt = 0; nt < 8; ++nt) {
            acc[nt] = __builtin_amdgcn_wmma_f32_16x16x32_f16(
                false, a.h, false, bf[nt].h, (short)0, acc[nt], false, false);
        }
    }

#pragma unroll
    for (int nt = 0; nt < 8; ++nt) {
        int col = nt * 16 + lrow;
        float bv = lb[col];
#pragma unroll
        for (int r = 0; r < 8; ++r) {
            int row = mBase + r + hi * 8;
            out[(size_t)row * DM + col] = acc[nt][r] + bv;
        }
    }
}

// ---------------------------------------------------------------------------
// Host side
// ---------------------------------------------------------------------------
extern "C" void kernel_launch(void* const* d_in, const int* in_sizes, int n_in,
                              void* d_out, int out_size, void* d_ws, size_t ws_size,
                              hipStream_t stream)
{
    const float* q    = (const float*)d_in[0];
    const float* k    = (const float*)d_in[1];
    const float* v    = (const float*)d_in[2];
    const float* ckw  = (const float*)d_in[3];
    const float* ckb  = (const float*)d_in[4];
    const float* cvw  = (const float*)d_in[5];
    const float* cvb  = (const float*)d_in[6];
    const float* lw   = (const float*)d_in[7];
    const float* lbv  = (const float*)d_in[8];
    const int*   isamp = (const int*)d_in[9];
    float* out = (float*)d_out;

    // Workspace layout (f32 segments first, then f16, then int)
    const size_t NBSD  = (size_t)B_DIM * BS_ELTS;      // 4,194,304 elements
    const size_t NPADT = (size_t)B_DIM * PAD_ROWS * 256;
    float* qr = (float*)d_ws;
    float* kr = qr + NBSD;
    float* kc = kr + NBSD;
    float* vc = kc + NBSD;
    _Float16* xh  = (_Float16*)(vc + NBSD);            // B*S*DM halves
    _Float16* whk = xh + NBSD;                         // 7*256*256
    _Float16* whv = whk + (size_t)KSZ * S_DIM * S_DIM;
    _Float16* wlh = whv + (size_t)KSZ * S_DIM * S_DIM; // 128*128
    _Float16* xtk = wlh + DM * DM;                     // B*134*256
    _Float16* xtv = xtk + NPADT;
    int* mtop = (int*)(xtv + NPADT);                   // B*H*32

    // 1. rotary (B*S*64 pairs)
    rotary_kernel<<<(B_DIM * S_DIM * 64) / 256, 256, 0, stream>>>(q, k, qr, kr);

    // 2. weight repack + padded transposed conv inputs (f16)
    convert_conv_w_kernel<<<(KSZ * S_DIM * S_DIM + 255) / 256, 256, 0, stream>>>(ckw, whk);
    convert_conv_w_kernel<<<(KSZ * S_DIM * S_DIM + 255) / 256, 256, 0, stream>>>(cvw, whv);
    convert_lin_kernel<<<(DM * DM + 255) / 256, 256, 0, stream>>>(lw, wlh);
    transpose_pad_kernel<<<(int)((NPADT + 255) / 256), 256, 0, stream>>>(kr, xtk);
    transpose_pad_kernel<<<(int)((NPADT + 255) / 256), 256, 0, stream>>>(v,  xtv);

    // 3. conv GEMMs
    conv_gemm_kernel<<<dim3(2, B_DIM), 256, 0, stream>>>(xtk, whk, ckb, kc);
    conv_gemm_kernel<<<dim3(2, B_DIM), 256, 0, stream>>>(xtv, whv, cvb, vc);

    // 4. sampled scoring + top-30
    topk_kernel<<<B_DIM * H_DIM, 256, 0, stream>>>(qr, kc, isamp, mtop);

    // 5. attention + context assembly (emits f16 activations)
    attn_kernel<<<B_DIM * H_DIM, 256, 0, stream>>>(qr, kc, vc, mtop, xh);

    // 6. output linear via WMMA (weights async-staged in LDS)
    linear_gemm_kernel<<<(B_DIM * S_DIM) / 128, 256, 0, stream>>>(xh, wlh, lbv, out);
}
